// TransformerConv_86766929314290
// MI455X (gfx1250) — matrix operand order repaired and checked by reading, not compiled, not run on hardware
//
#include <hip/hip_runtime.h>
#include <hip/hip_bf16.h>
#include <math.h>

// ---------------------------------------------------------------------------
// GAT layer for MI455X (gfx1250, wave32, WMMA).
//   Phase 1: three fp32 projections via V_WMMA_F32_16X16X4_F32
//   Phase 2: dst-CSR build (histogram -> scan -> scatter), atomic-free softmax
//   Phase 3: per-node segment softmax + weighted aggregation
// ---------------------------------------------------------------------------

#define N_NODES  50000
#define N_EDGES  800000
#define IN_F     128
#define HD       128          // HEADS * OUT_FEATS = 4 * 32
#define NEG_SLOPE 0.2f

typedef __attribute__((ext_vector_type(2)))  float    v2f;
typedef __attribute__((ext_vector_type(8)))  float    v8f;
typedef __attribute__((ext_vector_type(16))) _Float16 v16h;

// ---------------------------------------------------------------------------
// Phase 1: out[n, j] = sum_k feat[n,k] * W[j,k] + b[j]
// One block = 16 rows x 128 cols. 8 waves, each wave owns one 16x16 tile.
// blockIdx.y selects which of the three projections we compute.
// ---------------------------------------------------------------------------
__global__ __launch_bounds__(256) void gat_proj_wmma(
    const float* __restrict__ feat,
    const float* __restrict__ W0, const float* __restrict__ b0,
    const float* __restrict__ W1, const float* __restrict__ b1,
    const float* __restrict__ W2, const float* __restrict__ b2,
    float* __restrict__ o0, float* __restrict__ o1, float* __restrict__ o2)
{
    const float* W; const float* b; float* o;
    switch (blockIdx.y) {
        case 0:  W = W0; b = b0; o = o0; break;
        case 1:  W = W1; b = b1; o = o1; break;
        default: W = W2; b = b2; o = o2; break;
    }

    const int lane = threadIdx.x & 31;
    const int wave = threadIdx.x >> 5;      // 0..7 -> column tile
    const int m    = lane & 15;             // row (A) / col (B) within tile
    const int hi   = lane >> 4;             // half-wave select

    const int row = blockIdx.x * 16;        // 50000 = 16 * 3125, no tail
    const int col = wave * 16;

    const float* Arow = feat + (size_t)(row + m) * IN_F;   // A: 16x4 slabs
    const float* Brow = W    + (size_t)(col + m) * IN_F;   // B[k][n] = W[n][k]

    v8f acc = {};

#if __has_builtin(__builtin_amdgcn_wmma_f32_16x16x4_f32)
    // fp32 WMMA: A 16x4 layout -> lane m, K = k0 + 2*hi + {0,1}
    //            B 4x16 layout -> lane n=m, same K split
    #pragma unroll
    for (int k0 = 0; k0 < IN_F; k0 += 4) {
        v2f a, bf;
        a.x  = Arow[k0 + 2 * hi + 0];
        a.y  = Arow[k0 + 2 * hi + 1];
        bf.x = Brow[k0 + 2 * hi + 0];
        bf.y = Brow[k0 + 2 * hi + 1];
        acc = __builtin_amdgcn_wmma_f32_16x16x4_f32(
                  false, a, false, bf, (short)0, acc, false, false);
    }
#else
    // Fallback: f16 inputs, f32 accumulate (still emits v_wmma_*)
    #pragma unroll
    for (int k0 = 0; k0 < IN_F; k0 += 32) {
        v16h a, bb;
        #pragma unroll
        for (int j = 0; j < 16; ++j) {
            int ka = j + hi * 16;                           // contiguous k = 16*hi + j (B matches)
            a[j]  = (_Float16)Arow[k0 + ka];
            int kb = j + hi * 16;                           // contiguous k, same order as A
            bb[j] = (_Float16)Brow[k0 + kb];
        }
        acc = __builtin_amdgcn_wmma_f32_16x16x32_f16(
                  false, a, false, bb, (short)0, acc, false, false);
    }
#endif

    const float bias = b[col + m];
    #pragma unroll
    for (int r = 0; r < 8; ++r) {
        // C/D layout: VGPR r holds M = r + 8*hi, N = m
        int orow = row + r + 8 * hi;
        o[(size_t)orow * HD + col + m] = acc[r] + bias;
    }
}

// ---------------------------------------------------------------------------
// Phase 2 helpers: CSR over dst
// ---------------------------------------------------------------------------
__global__ void zero_i32(int* p, int n) {
    int i = blockIdx.x * blockDim.x + threadIdx.x;
    if (i < n) p[i] = 0;
}

__global__ void count_deg(const int* __restrict__ dst, int* __restrict__ counts, int e) {
    int i = blockIdx.x * blockDim.x + threadIdx.x;
    if (i < e) atomicAdd(&counts[dst[i]], 1);
}

// Single-block chunked exclusive scan, in place. n = N_NODES+1 (last slot 0
// before scan -> becomes total edge count after scan).
__global__ __launch_bounds__(1024) void scan_excl(int* data, int n) {
    __shared__ int sh[1024];
    __shared__ int carry;
    if (threadIdx.x == 0) carry = 0;
    __syncthreads();
    for (int base = 0; base < n; base += 1024) {
        int i = base + (int)threadIdx.x;
        int v = (i < n) ? data[i] : 0;
        sh[threadIdx.x] = v;
        __syncthreads();
        for (int off = 1; off < 1024; off <<= 1) {
            int t = (threadIdx.x >= (unsigned)off) ? sh[threadIdx.x - off] : 0;
            __syncthreads();
            sh[threadIdx.x] += t;
            __syncthreads();
        }
        int excl = sh[threadIdx.x] - v + carry;
        if (i < n) data[i] = excl;
        int chunk_total = sh[1023];
        __syncthreads();
        if (threadIdx.x == 0) carry += chunk_total;
        __syncthreads();
    }
}

__global__ void copy_i32(const int* __restrict__ a, int* __restrict__ b, int n) {
    int i = blockIdx.x * blockDim.x + threadIdx.x;
    if (i < n) b[i] = a[i];
}

__global__ void scatter_src(const int* __restrict__ src, const int* __restrict__ dst,
                            int* __restrict__ cursor, int* __restrict__ csr_src, int e) {
    int i = blockIdx.x * blockDim.x + threadIdx.x;
    if (i < e) {
        int p = atomicAdd(&cursor[dst[i]], 1);
        csr_src[p] = src[i];
    }
}

// ---------------------------------------------------------------------------
// Phase 3: per-node segment softmax + aggregation. One block per dst node,
// 128 threads: lane t owns element (h,d) = t of the HD=128 feature vector.
// Two passes over this node's incoming edges (max, then exp/accumulate).
// Atomic-free; all gathers are 512B coalesced and L2-resident.
// ---------------------------------------------------------------------------
__global__ __launch_bounds__(128) void gat_attn(
    const float* __restrict__ f_src, const float* __restrict__ f_dst,
    const float* __restrict__ f_v,
    const int* __restrict__ offsets, const int* __restrict__ csr_src,
    float* __restrict__ out)
{
    const int v  = blockIdx.x;
    const int hd = threadIdx.x;
    const int beg = offsets[v];
    const int end = offsets[v + 1];

    if (beg == end) {                       // no incoming edges -> zeros
        out[(size_t)v * HD + hd] = 0.0f;
        return;
    }

    const float fd = f_dst[(size_t)v * HD + hd];

    // Pass 1: per-element max of leaky_relu(f_src[s] + f_dst[v])
    float mx = -3.402823466e38f;
    for (int k = beg; k < end; ++k) {
        int s = csr_src[k];                 // same address across wave -> broadcast
        float a = f_src[(size_t)s * HD + hd] + fd;
        a = (a >= 0.0f) ? a : NEG_SLOPE * a;
        mx = fmaxf(mx, a);
    }

    // Pass 2: softmax numerator/denominator fused with value aggregation
    float num = 0.0f, den = 0.0f;
    for (int k = beg; k < end; ++k) {
        int s = csr_src[k];
        float a = f_src[(size_t)s * HD + hd] + fd;
        a = (a >= 0.0f) ? a : NEG_SLOPE * a;
        float e = __expf(a - mx);
        den += e;
        num += f_v[(size_t)s * HD + hd] * e;
    }

    out[(size_t)v * HD + hd] = num / den;
}

// ---------------------------------------------------------------------------
// Launch
// ---------------------------------------------------------------------------
extern "C" void kernel_launch(void* const* d_in, const int* in_sizes, int n_in,
                              void* d_out, int out_size, void* d_ws, size_t ws_size,
                              hipStream_t stream)
{
    const float* feat  = (const float*)d_in[0];
    const float* W_src = (const float*)d_in[1];
    const float* b_src = (const float*)d_in[2];
    const float* W_dst = (const float*)d_in[3];
    const float* b_dst = (const float*)d_in[4];
    const float* W_v   = (const float*)d_in[5];
    const float* b_v   = (const float*)d_in[6];
    const int*   src   = (const int*)d_in[7];
    const int*   dst   = (const int*)d_in[8];
    float*       out   = (float*)d_out;

    // workspace layout: 3 projected tables, CSR offsets/cursor/src (~80.4 MB)
    float* f_src   = (float*)d_ws;
    float* f_dst   = f_src + (size_t)N_NODES * HD;
    float* f_v     = f_dst + (size_t)N_NODES * HD;
    int*   offsets = (int*)(f_v + (size_t)N_NODES * HD);   // N+1
    int*   cursor  = offsets + (N_NODES + 1);              // N
    int*   csr     = cursor + N_NODES;                     // E

    // Phase 1: projections (WMMA). grid = (3125 row-tiles, 3 projections)
    gat_proj_wmma<<<dim3(N_NODES / 16, 3), 256, 0, stream>>>(
        feat, W_src, b_src, W_dst, b_dst, W_v, b_v, f_src, f_dst, f_v);

    // Phase 2: CSR build over dst
    zero_i32<<<(N_NODES + 1 + 255) / 256, 256, 0, stream>>>(offsets, N_NODES + 1);
    count_deg<<<(N_EDGES + 255) / 256, 256, 0, stream>>>(dst, offsets, N_EDGES);
    scan_excl<<<1, 1024, 0, stream>>>(offsets, N_NODES + 1);
    copy_i32<<<(N_NODES + 255) / 256, 256, 0, stream>>>(offsets, cursor, N_NODES);
    scatter_src<<<(N_EDGES + 255) / 256, 256, 0, stream>>>(src, dst, cursor, csr, N_EDGES);

    // Phase 3: segment softmax + aggregation, atomic-free
    gat_attn<<<N_NODES, 128, 0, stream>>>(f_src, f_dst, f_v, offsets, csr, out);
}
